// AttentionLayer_88931592831372
// MI455X (gfx1250) — matrix operand order repaired and checked by reading, not compiled, not run on hardware
//
#include <hip/hip_runtime.h>
#include <math.h>

// Shapes (fixed by the reference)
#define BB 32
#define TT 2048
#define DD 512
#define TC 64      // t-rows per workgroup in the main kernel
#define KSTEP 32   // K panel per WMMA accumulation step
#define NKS (DD / KSTEP)   // 16 K-steps

typedef __attribute__((ext_vector_type(8)))  float        v8f;
typedef __attribute__((ext_vector_type(4)))  __bf16       v4bf;
typedef __attribute__((ext_vector_type(8)))  __bf16       v8bf;
typedef __attribute__((ext_vector_type(16))) __bf16       v16bf;
typedef __attribute__((ext_vector_type(4)))  unsigned int v4u;
typedef __attribute__((ext_vector_type(8)))  int          v8i;
typedef __attribute__((ext_vector_type(4)))  int          v4i;

// ---------------------------------------------------------------------------
// Kernel 0: zero the o (B,D) and rowsum (B) accumulators in workspace.
// ---------------------------------------------------------------------------
__global__ void attn_init(float* __restrict__ o, float* __restrict__ rowsum) {
    int g = blockIdx.x * blockDim.x + threadIdx.x;   // 64*256 = 16384 = B*D
    if (g < BB * DD) o[g] = 0.0f;
    if (g < BB)      rowsum[g] = 0.0f;
}

// ---------------------------------------------------------------------------
// Kernel 1: pack W_h (f32, k-major) into bf16 K-panel-major layout:
//   Wt[((ks*512 + n)*32) + kk] = bf16(W_h[(ks*32+kk)*512 + n])
// Each 32KB panel is exactly the LDS image the main loop consumes -> the
// main kernel can DMA it with one TDM tensor_load_to_lds per K-step.
// One-time ~1.5MB of traffic, L2-resident thereafter.
// ---------------------------------------------------------------------------
__global__ void attn_prep_wt(const float* __restrict__ Wh, __bf16* __restrict__ Wt) {
    int g  = blockIdx.x * blockDim.x + threadIdx.x;  // 0 .. 262143
    int ks = g >> 14;          // K panel
    int r  = g & 16383;
    int n  = r >> 5;           // output column
    int kk = r & 31;           // k within panel
    Wt[g] = (__bf16)Wh[(ks * KSTEP + kk) * DD + n];
}

// ---------------------------------------------------------------------------
// Kernel 2: s1 = x0 @ W_a   (32x512 @ 512x512, ~17 MFLOP -> plain f32 FMA)
// ---------------------------------------------------------------------------
__global__ void attn_s1(const float* __restrict__ x0,
                        const float* __restrict__ Wa,
                        float* __restrict__ s1) {
    int g = blockIdx.x * blockDim.x + threadIdx.x;   // 16384 outputs
    int b = g >> 9;
    int d = g & (DD - 1);
    float acc = 0.0f;
    const float* x0r = x0 + b * DD;
    for (int k = 0; k < DD; ++k)
        acc = fmaf(x0r[k], Wa[k * DD + d], acc);
    s1[g] = acc;
}

// ---------------------------------------------------------------------------
// Kernel 3 (main): fused  s2 = x1@W_h -> tanh(s1+s2) -> sum_t V_a[t]*(),
// plus rowsum[b] = sum_{t,i} x1[b,t,i], reading x1 exactly once.
// bf16 WMMA (V_WMMA_F32_16X16X32_BF16), f32 accumulate.
// W panel staged via the Tensor Data Mover (tensor_load_to_lds).
// Grid: B * (T/TC) = 1024 workgroups of 512 threads (16 waves).
// ---------------------------------------------------------------------------
__launch_bounds__(512, 1)
__global__ void attn_main(const float* __restrict__ x1,
                          const __bf16* __restrict__ Wt,   // packed bf16 panels
                          const float* __restrict__ Va,
                          const float* __restrict__ s1,
                          float* __restrict__ o,
                          float* __restrict__ rowsum) {
    __shared__ __align__(128) __bf16 ldsA[TC * KSTEP];   // x1 panel [row][k], 4 KB
    __shared__ __align__(128) __bf16 ldsW[DD * KSTEP];   // W panel  [n][k], 32 KB

    const int tid  = threadIdx.x;
    const int lane = tid & 31;
    const int hi   = lane >> 4;       // WMMA half-wave split
    const int lr   = lane & 15;
    const int wave = tid >> 5;
    const int sWave = __builtin_amdgcn_readfirstlane(wave);  // scalar wave id

    const int b   = blockIdx.x >> 5;  // 32 t-chunks per batch
    const int tch = blockIdx.x & 31;
    const int t0  = tch * TC;

    const float* x1b = x1 + (size_t)b * TT * DD + (size_t)t0 * DD;

    // TDM constants: LDS byte offset of ldsW (generic LDS ptr low 32 bits) and
    // global base of the packed W table.
    const unsigned ldsWoff = (unsigned)(uintptr_t)(&ldsW[0]);
    const unsigned long long wtBase = (unsigned long long)(uintptr_t)Wt;

    // accumulators: 4 M-tiles x 2 N-tiles, 16x16 f32 each
    v8f acc[4][2];
    #pragma unroll
    for (int m = 0; m < 4; ++m)
        #pragma unroll
        for (int nt = 0; nt < 2; ++nt)
            #pragma unroll
            for (int i = 0; i < 8; ++i)
                acc[m][nt][i] = 0.0f;

    float xsum = 0.0f;

    // x1 staging coords (fixed across K-steps)
    const int srow = tid >> 3;   // 64 rows, 8 threads/row
    const int sc4  = tid & 7;    // float4 slot in the 32-col panel

    for (int ks = 0; ks < NKS; ++ks) {
        const int k0 = ks * KSTEP;
        __syncthreads();

        // ---- W panel: one TDM DMA (32 KB, contiguous -> contiguous) --------
        if (sWave == 0) {
            const unsigned long long ga =
                wtBase + (unsigned long long)ks * (DD * KSTEP * 2);
            v4u g0;
            g0[0] = 1u;                                   // count=1 (valid user D#)
            g0[1] = ldsWoff;                              // lds_addr (bytes)
            g0[2] = (unsigned)(ga & 0xFFFFFFFFu);         // global_addr[31:0]
            g0[3] = (unsigned)((ga >> 32) & 0x01FFFFFFu)  // global_addr[56:32]
                    | 0x80000000u;                        // type=2 ("image")
            v8i g1;
            g1[0] = 0x00020000;                           // data_size=2 (4B)
            g1[1] = (int)(0x2000u << 16);                 // tensor_dim0 = 8192 dwords (lo16)
            g1[2] = 0x00010000;                           // tensor_dim0 hi=0; tensor_dim1=1
            g1[3] = (int)(0x2000u << 16);                 // tile_dim0 = 8192 dwords
            g1[4] = 0;                                    // tile_dim1/2 unused
            g1[5] = 0x2000;                               // tensor_dim0_stride = 8192
            g1[6] = 0;
            g1[7] = 0;
            const v4i z4 = {0, 0, 0, 0};                  // groups 2/3 unused (<=2D)
            const v8i z8 = {0, 0, 0, 0, 0, 0, 0, 0};
            __builtin_amdgcn_tensor_load_to_lds(g0, g1, z4, z4, z8, 0);
        }

        // ---- x1 panel: 64x32 f32 -> bf16 LDS (packed b64 store) + rowsum ---
        {
            const float4 v = *(const float4*)(x1b + (size_t)srow * DD + k0 + sc4 * 4);
            xsum += v.x + v.y + v.z + v.w;
            v4bf p;
            p[0] = (__bf16)v.x; p[1] = (__bf16)v.y;
            p[2] = (__bf16)v.z; p[3] = (__bf16)v.w;
            *(v4bf*)(&ldsA[srow * KSTEP + sc4 * 4]) = p;   // ds_store_b64
            if (ks + 1 < NKS)   // speculative prefetch of next x1 panel
                __builtin_prefetch(x1b + (size_t)srow * DD + k0 + KSTEP + sc4 * 4, 0, 1);
        }

        if (sWave == 0)
            __builtin_amdgcn_s_wait_tensorcnt(0);          // TDM LDS writes done
        __syncthreads();

        // ---- B fragments: lane = col (lane&15), K half = 16*(lane>>4) ------
        v16bf bfrag[2];
        #pragma unroll
        for (int nt = 0; nt < 2; ++nt) {
            const __bf16* p = &ldsW[(wave * 32 + nt * 16 + lr) * KSTEP + hi * 16];
            bfrag[nt] = *(const v16bf*)p;                  // 2x ds_load_b128
        }

        // ---- A fragments + WMMA -------------------------------------------
        #pragma unroll
        for (int m = 0; m < 4; ++m) {
            // A 16x32 fragment, contiguous k order as in B: lane M = lane&15;
            // lanes 0-15: K 0..15 ; lanes 16-31: K 16..31
            const __bf16* pa = &ldsA[(m * 16 + lr) * KSTEP + hi * 16];
            v8bf alo = *(const v8bf*)pa;
            v8bf ahi = *(const v8bf*)(pa + 8);
            v16bf afrag;
            #pragma unroll
            for (int i = 0; i < 8; ++i) { afrag[i] = alo[i]; afrag[8 + i] = ahi[i]; }
            #pragma unroll
            for (int nt = 0; nt < 2; ++nt)
                acc[m][nt] = __builtin_amdgcn_wmma_f32_16x16x32_bf16(
                    false, afrag, false, bfrag[nt], (short)0, acc[m][nt], false, false);
        }
    }

    // ---- epilogue: tanh(s2 + s1) * V_a[t], reduce over local t-rows --------
    // C layout: VGPR i -> M = i + 8*hi ; N = lane&15
    float va[4][8];
    #pragma unroll
    for (int m = 0; m < 4; ++m)
        #pragma unroll
        for (int i = 0; i < 8; ++i)
            va[m][i] = Va[t0 + m * 16 + hi * 8 + i];

    #pragma unroll
    for (int nt = 0; nt < 2; ++nt) {
        const int col = wave * 32 + nt * 16 + lr;
        const float s1v = s1[b * DD + col];
        float csum = 0.0f;
        #pragma unroll
        for (int m = 0; m < 4; ++m)
            #pragma unroll
            for (int i = 0; i < 8; ++i)
                csum += tanhf(acc[m][nt][i] + s1v) * va[m][i];
        csum += __shfl_xor(csum, 16, 32);        // fold the two lane-halves
        if (hi == 0)
            atomicAdd(&o[b * DD + col], csum);   // global_atomic_add_f32
    }

    // ---- rowsum[b]: wave-reduce, one atomic per wave -----------------------
    #pragma unroll
    for (int off = 16; off > 0; off >>= 1)
        xsum += __shfl_xor(xsum, off, 32);
    if (lane == 0)
        atomicAdd(&rowsum[b], xsum);
}

// ---------------------------------------------------------------------------
// Kernel 4: out[b,d] = softmax_D(o[b,:])[d] * rowsum[b]
// ---------------------------------------------------------------------------
__global__ void attn_softmax(const float* __restrict__ o,
                             const float* __restrict__ rowsum,
                             float* __restrict__ out) {
    const int b    = blockIdx.x;
    const int tid  = threadIdx.x;    // 256 threads, 2 cols each
    const int lane = tid & 31;
    const int wave = tid >> 5;
    __shared__ float sred[8];

    float v0 = o[b * DD + tid];
    float v1 = o[b * DD + 256 + tid];

    float m = fmaxf(v0, v1);
    #pragma unroll
    for (int off = 16; off > 0; off >>= 1)
        m = fmaxf(m, __shfl_xor(m, off, 32));
    if (lane == 0) sred[wave] = m;
    __syncthreads();
    float bm = sred[0];
    #pragma unroll
    for (int i = 1; i < 8; ++i) bm = fmaxf(bm, sred[i]);
    __syncthreads();

    float e0 = __expf(v0 - bm);
    float e1 = __expf(v1 - bm);
    float s = e0 + e1;
    #pragma unroll
    for (int off = 16; off > 0; off >>= 1)
        s += __shfl_xor(s, off, 32);
    if (lane == 0) sred[wave] = s;
    __syncthreads();
    float tot = 0.0f;
    #pragma unroll
    for (int i = 0; i < 8; ++i) tot += sred[i];

    const float scale = rowsum[b] / tot;
    out[b * DD + tid]       = e0 * scale;
    out[b * DD + 256 + tid] = e1 * scale;
}

// ---------------------------------------------------------------------------
// Host launcher
// Inputs (setup_inputs order): 0=x0 (B,D) 1=x1 (B,T,D) 2=W_a (D,D)
//                              3=W_h (D,D) 4=V_a (1,T);  out: (B,D) f32
// ---------------------------------------------------------------------------
extern "C" void kernel_launch(void* const* d_in, const int* in_sizes, int n_in,
                              void* d_out, int out_size, void* d_ws, size_t ws_size,
                              hipStream_t stream) {
    const float* x0 = (const float*)d_in[0];
    const float* x1 = (const float*)d_in[1];
    const float* Wa = (const float*)d_in[2];
    const float* Wh = (const float*)d_in[3];
    const float* Va = (const float*)d_in[4];
    float* out = (float*)d_out;

    // workspace: Wt (512KB bf16 packed panels) | s1 (B*D f32) | o (B*D) | rowsum (B)
    __bf16* Wt    = (__bf16*)d_ws;
    float* s1     = (float*)((char*)d_ws + (size_t)DD * DD * sizeof(__bf16));
    float* o      = s1 + BB * DD;
    float* rowsum = o + BB * DD;

    attn_init<<<64, 256, 0, stream>>>(o, rowsum);
    attn_prep_wt<<<(DD * DD) / 256, 256, 0, stream>>>(Wh, Wt);
    attn_s1<<<(BB * DD) / 256, 256, 0, stream>>>(x0, Wa, s1);
    attn_main<<<BB * (TT / TC), 512, 0, stream>>>(x1, Wt, Va, s1, o, rowsum);
    attn_softmax<<<BB, 256, 0, stream>>>(o, rowsum, out);
}
